// BotRGCN_64381559767213
// MI455X (gfx1250) — compile-verified, matrix operand order checked by reading, not run on hardware
//
#include <hip/hip_runtime.h>

typedef __attribute__((ext_vector_type(16))) __bf16 v16bf;
typedef __attribute__((ext_vector_type(8)))  __bf16 v8bf;
typedef __attribute__((ext_vector_type(8)))  float  v8f;

#define D_   128
#define KDES 768

__device__ __forceinline__ float lrelu_f(float v) { return v > 0.f ? v : 0.01f * v; }

// Native CDNA5 f32 atomic add, device scope (cross-WGP coherent, L2-resident RMW).
__device__ __forceinline__ void atomAddF(float* p, float v) {
    asm volatile("global_atomic_add_f32 %0, %1, off scope:SCOPE_DEV"
                 : : "v"(p), "v"(v) : "memory");
}

#define WMMA_BF16(A, B, C) \
    __builtin_amdgcn_wmma_f32_16x16x32_bf16(false, (A), false, (B), (short)0, (C), false, false)

// ---- Fragment loaders (ISA 7.12.2 layouts) ----
// A fragment from bf16 row-major [rows x ld]: two 16B chunks per lane.
__device__ __forceinline__ v16bf load_a_bf(const __bf16* __restrict__ src,
                                           int row0, int ld, int k0, int lane) {
    int m = lane & 15, kh = lane >> 4;
    const __bf16* p = src + (size_t)(row0 + m) * ld + k0 + kh * 8;
    v8bf c0 = *(const v8bf*)p;         // K = k0+kh*8 .. +7
    v8bf c1 = *(const v8bf*)(p + 16);  // K = k0+16+kh*8 .. +7
    return __builtin_shufflevector(c0, c1, 0, 1, 2, 3, 4, 5, 6, 7,
                                   8, 9, 10, 11, 12, 13, 14, 15);
}

// A fragment from LDS-staged bf16 tile [16 x ldk] (ds_load_b128 x2).
__device__ __forceinline__ v16bf lds_a_bf(const __bf16* base, int ldk, int k0, int lane) {
    int m = lane & 15, kh = lane >> 4;
    const __bf16* p = base + m * ldk + k0 + kh * 8;
    v8bf c0 = *(const v8bf*)p;
    v8bf c1 = *(const v8bf*)(p + 16);
    return __builtin_shufflevector(c0, c1, 0, 1, 2, 3, 4, 5, 6, 7,
                                   8, 9, 10, 11, 12, 13, 14, 15);
}

// B fragment from pre-packed fragment-ordered weight:
// layout [K/32 kblk][8 colTile][2 chunk][32 lane][8 bf16] -> two coalesced b128 loads.
__device__ __forceinline__ v16bf load_b_packed(const __bf16* __restrict__ packed,
                                               int k0, int colTile, int lane) {
    const __bf16* p = packed + ((((size_t)(k0 >> 5) * 8 + colTile) * 2 * 32) + lane) * 8;
    v8bf c0 = *(const v8bf*)p;           // chunk 0
    v8bf c1 = *(const v8bf*)(p + 256);   // chunk 1 (+32 lanes * 8 elems)
    return __builtin_shufflevector(c0, c1, 0, 1, 2, 3, 4, 5, 6, 7,
                                   8, 9, 10, 11, 12, 13, 14, 15);
}

// ---- One-shot: pack f32 row-major [K x 128] weight into bf16 WMMA-B fragment order ----
__global__ void pack_b_kernel(const float* __restrict__ w, __bf16* __restrict__ packed, int K) {
    int idx = blockIdx.x * blockDim.x + threadIdx.x; // one thread per (kblk, colTile, chunk, lane)
    int total = (K / 32) * 8 * 2 * 32;
    if (idx >= total) return;
    int lane  = idx & 31;
    int chunk = (idx >> 5) & 1;
    int c     = (idx >> 6) & 7;
    int kb0   = idx >> 9;
    int n = lane & 15, kh = lane >> 4;
    int col = c * 16 + n;
    __bf16* dst = packed + (size_t)idx * 8;
#pragma unroll
    for (int j = 0; j < 4; ++j) {
        int kb = kb0 * 32 + chunk * 16 + kh * 8 + 2 * j;
        dst[2 * j]     = (__bf16)w[(size_t)kb * D_ + col];
        dst[2 * j + 1] = (__bf16)w[(size_t)(kb + 1) * D_ + col];
    }
}

// ---- Kernel 1: fused multimodal projection -> xbf[N x 128] (bf16) ----
__global__ void __launch_bounds__(256)
proj_kernel(const float* __restrict__ des, const float* __restrict__ tw,
            const float* __restrict__ num, const float* __restrict__ cat,
            const __bf16* __restrict__ des_wp, const float* __restrict__ des_b,
            const __bf16* __restrict__ tw_wp,  const float* __restrict__ tw_b,
            const float* __restrict__ num_w, const float* __restrict__ num_b,
            const float* __restrict__ cat_w, const float* __restrict__ cat_b,
            __bf16* __restrict__ xbf) {
    __shared__ __bf16 smem[2 * 16 * KDES];  // 48 KB: [mat][row][k]

    int t    = threadIdx.x;
    int lane = t & 31;
    int wave = t >> 5;         // col tile 0..7
    int row0 = blockIdx.x * 16;

    // Stage + convert A tiles once: (mat, row) owned by groups of 8 threads.
    {
        int mat = t >> 7;             // 0=des, 1=tweets
        int row = (t >> 3) & 15;
        int lp  = t & 7;
        const float* src = (mat ? tw : des) + (size_t)(row0 + row) * KDES;
        __bf16* dst = smem + (mat * 16 + row) * KDES;
#pragma unroll 4
        for (int kp = lp; kp < KDES / 2; kp += 8) {
            float2 f = *(const float2*)(src + 2 * kp);
            union { unsigned u; __bf16 h[2]; } pk;
            pk.h[0] = (__bf16)f.x;
            pk.h[1] = (__bf16)f.y;
            *(unsigned*)(dst + 2 * kp) = pk.u;
        }
    }
    __syncthreads();

    v8f accd = {}; v8f acct = {};
    for (int k0 = 0; k0 < KDES; k0 += 32) {
        v16bf ad = lds_a_bf(smem, KDES, k0, lane);
        v16bf bd = load_b_packed(des_wp, k0, wave, lane);
        accd = WMMA_BF16(ad, bd, accd);
        v16bf at = lds_a_bf(smem + 16 * KDES, KDES, k0, lane);
        v16bf bt = load_b_packed(tw_wp, k0, wave, lane);
        acct = WMMA_BF16(at, bt, acct);
    }

    int n  = wave * 16 + (lane & 15);
    int kh = lane >> 4;
    float bnd = des_b[n], bnt = tw_b[n], bnn = num_b[n], bnc = cat_b[n];
#pragma unroll
    for (int v = 0; v < 8; ++v) {
        int row = row0 + v + 8 * kh;
        float sn = bnn;
#pragma unroll
        for (int k = 0; k < 5; ++k) sn += num[row * 5 + k] * num_w[k * D_ + n];
        float sc = bnc;
#pragma unroll
        for (int k = 0; k < 6; ++k) sc += cat[row * 6 + k] * cat_w[k * D_ + n];
        float r = lrelu_f(accd[v] + bnd) + lrelu_f(acct[v] + bnt) + lrelu_f(sn) + lrelu_f(sc);
        xbf[(size_t)row * D_ + n] = (__bf16)r;
    }
}

// ---- Kernel 2: per-layer node transforms: y0 = x@W0, y1 = x@W1, out = x@Wroot + b ----
__global__ void __launch_bounds__(256)
transform_kernel(const __bf16* __restrict__ xbf,
                 const __bf16* __restrict__ w0p, const __bf16* __restrict__ w1p,
                 const __bf16* __restrict__ wrp, const float* __restrict__ bias,
                 float* __restrict__ y0, float* __restrict__ y1,
                 float* __restrict__ out) {
    int lane = threadIdx.x & 31;
    int wave = threadIdx.x >> 5;
    int row0 = blockIdx.x * 16;

    v8f a0 = {}, a1 = {}, ar = {};
    for (int k0 = 0; k0 < D_; k0 += 32) {
        v16bf af = load_a_bf(xbf, row0, D_, k0, lane);
        v16bf b0 = load_b_packed(w0p, k0, wave, lane);
        a0 = WMMA_BF16(af, b0, a0);
        v16bf b1 = load_b_packed(w1p, k0, wave, lane);
        a1 = WMMA_BF16(af, b1, a1);
        v16bf br = load_b_packed(wrp, k0, wave, lane);
        ar = WMMA_BF16(af, br, ar);
    }

    int n  = wave * 16 + (lane & 15);
    int kh = lane >> 4;
    float bn = bias[n];
#pragma unroll
    for (int v = 0; v < 8; ++v) {
        size_t o = (size_t)(row0 + v + 8 * kh) * D_ + n;
        y0[o]  = a0[v];
        y1[o]  = a1[v];
        out[o] = ar[v] + bn;
    }
}

// ---- Degree counting / inverse ----
__global__ void zero_kernel(float* __restrict__ p, int n) {
    int i = blockIdx.x * blockDim.x + threadIdx.x;
    if (i < n) p[i] = 0.f;
}

__global__ void count_kernel(const int* __restrict__ ei, const int* __restrict__ et,
                             float* __restrict__ cnt, int E, int N) {
    int e = blockIdx.x * blockDim.x + threadIdx.x;
    if (e >= E) return;
    atomAddF(&cnt[(size_t)et[e] * N + ei[E + e]], 1.0f);
}

__global__ void invert_kernel(float* __restrict__ cnt, int n) {
    int i = blockIdx.x * blockDim.x + threadIdx.x;
    if (i < n) cnt[i] = 1.0f / fmaxf(cnt[i], 1.0f);
}

// ---- Edge scatter: one wave per edge, float4 per lane, native f32 atomics ----
__global__ void __launch_bounds__(256)
scatter_kernel(const int* __restrict__ ei, const int* __restrict__ et,
               const float* __restrict__ y0, const float* __restrict__ y1,
               const float* __restrict__ invcnt, float* __restrict__ out,
               int E, int N) {
    int e = blockIdx.x * (blockDim.x >> 5) + (threadIdx.x >> 5);
    if (e >= E) return;
    int lane = threadIdx.x & 31;
    int s = ei[e];
    int d = ei[E + e];
    int r = et[e];
    const float* y = r ? y1 : y0;
    float inv = invcnt[(size_t)r * N + d];
    float4 v = ((const float4*)(y + (size_t)s * D_))[lane];
    float* o = out + (size_t)d * D_ + lane * 4;
    atomAddF(o + 0, v.x * inv);
    atomAddF(o + 1, v.y * inv);
    atomAddF(o + 2, v.z * inv);
    atomAddF(o + 3, v.w * inv);
}

// leaky-relu + truncate to bf16 for the next layer's A operand
__global__ void lrelu_bf_kernel(const float* __restrict__ in, __bf16* __restrict__ outp, int n) {
    int i = blockIdx.x * blockDim.x + threadIdx.x;
    if (i < n) outp[i] = (__bf16)lrelu_f(in[i]);
}

// ---- MLP head ----
__global__ void __launch_bounds__(256)
mlp1_kernel(const __bf16* __restrict__ xbf, const __bf16* __restrict__ w1p,
            const float* __restrict__ b1, float* __restrict__ h) {
    int lane = threadIdx.x & 31;
    int wave = threadIdx.x >> 5;
    int row0 = blockIdx.x * 16;

    v8f acc = {};
    for (int k0 = 0; k0 < D_; k0 += 32) {
        v16bf af = load_a_bf(xbf, row0, D_, k0, lane);
        v16bf bf = load_b_packed(w1p, k0, wave, lane);
        acc = WMMA_BF16(af, bf, acc);
    }
    int n  = wave * 16 + (lane & 15);
    int kh = lane >> 4;
    float bn = b1[n];
#pragma unroll
    for (int v = 0; v < 8; ++v) {
        float t = acc[v] + bn;
        h[(size_t)(row0 + v + 8 * kh) * D_ + n] = t > 0.f ? t : 0.f;
    }
}

__global__ void mlp2_kernel(const float* __restrict__ h, const float* __restrict__ w2,
                            const float* __restrict__ b2, float* __restrict__ out, int N) {
    int idx = blockIdx.x * blockDim.x + threadIdx.x;
    if (idx >= N * 2) return;
    int i = idx >> 1, j = idx & 1;
    float s = b2[j];
    const float* hr = h + (size_t)i * D_;
#pragma unroll 8
    for (int k = 0; k < D_; ++k) s += hr[k] * w2[k * 2 + j];
    out[idx] = s;
}

extern "C" void kernel_launch(void* const* d_in, const int* in_sizes, int n_in,
                              void* d_out, int out_size, void* d_ws, size_t ws_size,
                              hipStream_t stream) {
    const float* des    = (const float*)d_in[0];
    const float* tweets = (const float*)d_in[1];
    const float* num    = (const float*)d_in[2];
    const float* cat    = (const float*)d_in[3];
    const int*   ei     = (const int*)d_in[4];   // [2, E]
    const int*   et     = (const int*)d_in[5];   // [E]
    const float* des_w  = (const float*)d_in[6];
    const float* des_b  = (const float*)d_in[7];
    const float* tw_w   = (const float*)d_in[8];
    const float* tw_b   = (const float*)d_in[9];
    const float* num_w  = (const float*)d_in[10];
    const float* num_b  = (const float*)d_in[11];
    const float* cat_w  = (const float*)d_in[12];
    const float* cat_b  = (const float*)d_in[13];
    const float* rel_w  = (const float*)d_in[14]; // [L,R,D,D]
    const float* root_w = (const float*)d_in[15]; // [L,D,D]
    const float* rgcn_b = (const float*)d_in[16]; // [L,D]
    const float* mlp_w1 = (const float*)d_in[17];
    const float* mlp_b1 = (const float*)d_in[18];
    const float* mlp_w2 = (const float*)d_in[19];
    const float* mlp_b2 = (const float*)d_in[20];

    const int N = in_sizes[0] / KDES;
    const int E = in_sizes[5];
    const size_t ND = (size_t)N * D_;

    // ---- workspace carve-up ----
    float* wsf  = (float*)d_ws;
    float* y0   = wsf;            // N*D f32
    float* y1   = wsf + ND;       // N*D f32
    float* obuf = wsf + 2 * ND;   // N*D f32
    float* cnt  = wsf + 3 * ND;   // 2*N f32 (inverse degrees)
    float* h    = y0;             // reuse after RGCN layers

    __bf16* bws    = (__bf16*)(wsf + 3 * ND + 2 * N);
    __bf16* xbf    = bws;                       // N*D
    __bf16* des_wp = xbf + ND;                  // 768*128 (packed)
    __bf16* tw_wp  = des_wp + KDES * D_;        // 768*128 (packed)
    __bf16* rel_wp = tw_wp + KDES * D_;         // 2*2*128*128 (packed)
    __bf16* root_wp= rel_wp + 4 * D_ * D_;      // 2*128*128 (packed)
    __bf16* mlp_wp = root_wp + 2 * D_ * D_;     // 128*128 (packed)

    const int rowTiles = N / 16;
    const int packBig  = (KDES / 32) * 8 * 2 * 32;  // threads per 768x128 pack
    const int packSm   = (D_ / 32) * 8 * 2 * 32;    // threads per 128x128 pack

    // 0) one-shot weight conversion + fragment-order packing (tiny)
    pack_b_kernel<<<(packBig + 255) / 256, 256, 0, stream>>>(des_w, des_wp, KDES);
    pack_b_kernel<<<(packBig + 255) / 256, 256, 0, stream>>>(tw_w, tw_wp, KDES);
    for (int i = 0; i < 4; ++i)
        pack_b_kernel<<<(packSm + 255) / 256, 256, 0, stream>>>(
            rel_w + (size_t)i * D_ * D_, rel_wp + (size_t)i * D_ * D_, D_);
    for (int i = 0; i < 2; ++i)
        pack_b_kernel<<<(packSm + 255) / 256, 256, 0, stream>>>(
            root_w + (size_t)i * D_ * D_, root_wp + (size_t)i * D_ * D_, D_);
    pack_b_kernel<<<(packSm + 255) / 256, 256, 0, stream>>>(mlp_w1, mlp_wp, D_);

    // 1) fused multimodal projection (LDS-staged bf16 A, packed B, WMMA f32 accum)
    proj_kernel<<<rowTiles, 256, 0, stream>>>(des, tweets, num, cat,
                                              des_wp, des_b, tw_wp, tw_b,
                                              num_w, num_b, cat_w, cat_b, xbf);

    // 2) per-relation in-degree reciprocals (graph is layer-invariant)
    zero_kernel<<<(2 * N + 255) / 256, 256, 0, stream>>>(cnt, 2 * N);
    count_kernel<<<(E + 255) / 256, 256, 0, stream>>>(ei, et, cnt, E, N);
    invert_kernel<<<(2 * N + 255) / 256, 256, 0, stream>>>(cnt, 2 * N);

    // 3) RGCN layers: node transforms first (N*D*D, not E*D*D FLOPs), then edge scatter.
    for (int l = 0; l < 2; ++l) {
        const __bf16* w0 = rel_wp + (size_t)l * 2 * D_ * D_;
        const __bf16* w1 = w0 + (size_t)D_ * D_;
        const __bf16* wr = root_wp + (size_t)l * D_ * D_;
        const float*  bl = rgcn_b + (size_t)l * D_;
        transform_kernel<<<rowTiles, 256, 0, stream>>>(xbf, w0, w1, wr, bl, y0, y1, obuf);
        scatter_kernel<<<(E + 7) / 8, 256, 0, stream>>>(ei, et, y0, y1, cnt, obuf, E, N);
        lrelu_bf_kernel<<<((int)ND + 255) / 256, 256, 0, stream>>>(obuf, xbf, (int)ND);
    }

    // 4) MLP head
    mlp1_kernel<<<rowTiles, 256, 0, stream>>>(xbf, mlp_wp, mlp_b1, h);
    mlp2_kernel<<<(N * 2 + 255) / 256, 256, 0, stream>>>(h, mlp_w2, mlp_b2, (float*)d_out, N);
}
